// L1Attn_8589934848
// MI455X (gfx1250) — compile-verified, matrix operand order checked by reading, not run on hardware
//
#include <hip/hip_runtime.h>
#include <hip/hip_bf16.h>

typedef _Float16 v2h  __attribute__((ext_vector_type(2)));
typedef _Float16 v16h __attribute__((ext_vector_type(16)));
typedef float    v8f  __attribute__((ext_vector_type(8)));

#define Bsz 8
#define Ssz 1024
#define Hsz 8
#define Dsz 64

// packed f16x2 subtract in a dword: v_pk_add_f16 with neg modifiers
__device__ __forceinline__ unsigned sub2(unsigned a, unsigned b) {
    v2h d = __builtin_bit_cast(v2h, a) - __builtin_bit_cast(v2h, b);
    return __builtin_bit_cast(unsigned, d);
}

__device__ __forceinline__ unsigned cvt2(float a, float b) {
    v2h r;
    r.x = (_Float16)a;
    r.y = (_Float16)b;
    return __builtin_bit_cast(unsigned, r);
}

union AV { v16h v; unsigned u[8]; };

__global__ void __launch_bounds__(256, 2)
l1attn_wmma_kernel(const float* __restrict__ q,
                   const float* __restrict__ k,
                   float* __restrict__ out) {
    // LDS: k tile [h][i][d] f16, q tile [h][j][d] f16, out stage [i][j][h] f32
    __shared__ __align__(16) unsigned short lds_k[Hsz * 16 * Dsz]; // 16KB
    __shared__ __align__(16) unsigned short lds_q[Hsz * 16 * Dsz]; // 16KB
    __shared__ __align__(16) float          stage[16 * 16 * Hsz];  // 8KB

    const int bx  = blockIdx.x;            // 0 .. 8*64*64-1
    const int b   = bx >> 12;
    const int rem = bx & 4095;
    const int it  = rem >> 6;              // key tile
    const int jt  = rem & 63;              // query tile
    const int tid = threadIdx.x;

    // ---- stage loads: f32 global -> f16 LDS (convert once) ----
    {
        const int rid    = tid >> 1;        // 0..127  -> (h, s_local)
        const int h      = rid >> 4;
        const int sl     = rid & 15;
        const int half64 = tid & 1;         // which 32-d half of the row

        const size_t koff = ((((size_t)b * Ssz + (size_t)(it * 16 + sl)) * Hsz + h) * Dsz) + half64 * 32;
        const size_t qoff = ((((size_t)b * Ssz + (size_t)(jt * 16 + sl)) * Hsz + h) * Dsz) + half64 * 32;
        const float4* kp = (const float4*)(k + koff);
        const float4* qp = (const float4*)(q + qoff);

        unsigned short* lk = &lds_k[(h * 16 + sl) * Dsz + half64 * 32];
        unsigned short* lq = &lds_q[(h * 16 + sl) * Dsz + half64 * 32];

        uint4 pk[4], pq[4];
        #pragma unroll
        for (int u = 0; u < 4; ++u) {
            float4 a = kp[2 * u], c = kp[2 * u + 1];
            pk[u] = make_uint4(cvt2(a.x, a.y), cvt2(a.z, a.w), cvt2(c.x, c.y), cvt2(c.z, c.w));
            float4 e = qp[2 * u], f = qp[2 * u + 1];
            pq[u] = make_uint4(cvt2(e.x, e.y), cvt2(e.z, e.w), cvt2(f.x, f.y), cvt2(f.z, f.w));
        }
        #pragma unroll
        for (int u = 0; u < 4; ++u) {
            ((uint4*)lk)[u] = pk[u];
            ((uint4*)lq)[u] = pq[u];
        }
    }
    __syncthreads();

    // ---- compute: wave w handles head h = w; one 16x16 (i x j) tile ----
    // Roles: A = all-ones (constant), B = abs-diffs.
    //   B operand, k order chosen here (A all-ones, any order works): lane owns column n = lane%16;
    //   lanes 0-15 hold K=0..15 (d = chunk*32 + 0..15),
    //   lanes 16-31 hold K=16..31 (d = chunk*32 + 16..31);
    //   dword v of the operand = K = 2v, 2v+1  -> 8 contiguous dwords/lane.
    //   D[m,n] = sum_K B[K,n] in every row m -> take acc[0] in each lane.
    const int wave  = tid >> 5;            // == h
    const int lane  = tid & 31;
    const int col   = lane & 15;           // query j column this lane owns
    const int halfl = lane >> 4;           // which K-half

    // Hoist this lane's q data: d = chunk*32 + halfl*16 + 0..15, chunks 0,1
    const unsigned short* qrow = &lds_q[(wave * 16 + col) * Dsz + halfl * 16];
    const uint4 q0a = ((const uint4*)qrow)[0];        // chunk0, d +0..7
    const uint4 q0b = ((const uint4*)qrow)[1];        // chunk0, d +8..15
    const uint4 q1a = ((const uint4*)(qrow + 32))[0]; // chunk1, d +0..7
    const uint4 q1b = ((const uint4*)(qrow + 32))[1]; // chunk1, d +8..15

    // Constant all-ones A matrix (f16 1.0 pairs), built once.
    AV ones;
    #pragma unroll
    for (int u = 0; u < 8; ++u) ones.u[u] = 0x3C003C00u;

    // key rows, software-pipelined: preload row g+1 before the WMMA pair
    const unsigned short* kbase = &lds_k[(wave * 16) * Dsz + halfl * 16];
    uint4 ka = ((const uint4*)kbase)[0];
    uint4 kb = ((const uint4*)kbase)[1];
    uint4 kc = ((const uint4*)(kbase + 32))[0];
    uint4 kd = ((const uint4*)(kbase + 32))[1];

    float* srow = &stage[col * Hsz + wave];

    #pragma unroll
    for (int g = 0; g < 16; ++g) {
        uint4 na, nb, nc, nd;
        if (g < 15) {
            const unsigned short* nk = kbase + (g + 1) * Dsz;
            na = ((const uint4*)nk)[0];
            nb = ((const uint4*)nk)[1];
            nc = ((const uint4*)(nk + 32))[0];
            nd = ((const uint4*)(nk + 32))[1];
        }

        // phase 1: 16 independent packed subtracts
        unsigned s0  = sub2(ka.x, q0a.x), s1  = sub2(ka.y, q0a.y);
        unsigned s2  = sub2(ka.z, q0a.z), s3  = sub2(ka.w, q0a.w);
        unsigned s4  = sub2(kb.x, q0b.x), s5  = sub2(kb.y, q0b.y);
        unsigned s6  = sub2(kb.z, q0b.z), s7  = sub2(kb.w, q0b.w);
        unsigned s8  = sub2(kc.x, q1a.x), s9  = sub2(kc.y, q1a.y);
        unsigned s10 = sub2(kc.z, q1a.z), s11 = sub2(kc.w, q1a.w);
        unsigned s12 = sub2(kd.x, q1b.x), s13 = sub2(kd.y, q1b.y);
        unsigned s14 = sub2(kd.z, q1b.z), s15 = sub2(kd.w, q1b.w);

        // phase 2: 16 independent abs (clear packed sign bits)
        AV B0, B1;
        B0.u[0] = s0  & 0x7fff7fffu;  B0.u[1] = s1  & 0x7fff7fffu;
        B0.u[2] = s2  & 0x7fff7fffu;  B0.u[3] = s3  & 0x7fff7fffu;
        B0.u[4] = s4  & 0x7fff7fffu;  B0.u[5] = s5  & 0x7fff7fffu;
        B0.u[6] = s6  & 0x7fff7fffu;  B0.u[7] = s7  & 0x7fff7fffu;
        B1.u[0] = s8  & 0x7fff7fffu;  B1.u[1] = s9  & 0x7fff7fffu;
        B1.u[2] = s10 & 0x7fff7fffu;  B1.u[3] = s11 & 0x7fff7fffu;
        B1.u[4] = s12 & 0x7fff7fffu;  B1.u[5] = s13 & 0x7fff7fffu;
        B1.u[6] = s14 & 0x7fff7fffu;  B1.u[7] = s15 & 0x7fff7fffu;

        v8f acc = {};
        acc = __builtin_amdgcn_wmma_f32_16x16x32_f16(false, ones.v, false, B0.v,
                                                     (short)0, acc, false, false);
        acc = __builtin_amdgcn_wmma_f32_16x16x32_f16(false, ones.v, false, B1.v,
                                                     (short)0, acc, false, false);

        // out[i=g][j=col]; both lane halves hold the identical value and write
        // the identical address (benign, keeps EXEC full - no divergence).
        srow[g * 16 * Hsz] = acc[0] * -0.125f;

        ka = na; kb = nb; kc = nc; kd = nd;
    }
    __syncthreads();

    // ---- coalesced store: per i-row, 16 j * 8 h = 512B contiguous ----
    {
        const int il  = tid >> 4;
        const int seg = tid & 15;          // 8 floats (32B) each
        const size_t obase = (((size_t)b * Ssz + (size_t)(it * 16 + il)) * Ssz
                              + (size_t)(jt * 16)) * Hsz + (size_t)(seg * 8);
        const float* sp = &stage[il * 128 + seg * 8];
        float4* op = (float4*)(out + obase);
        op[0] = ((const float4*)sp)[0];
        op[1] = ((const float4*)sp)[1];
    }
}

extern "C" void kernel_launch(void* const* d_in, const int* in_sizes, int n_in,
                              void* d_out, int out_size, void* d_ws, size_t ws_size,
                              hipStream_t stream) {
    (void)in_sizes; (void)n_in; (void)d_ws; (void)ws_size; (void)out_size;
    const float* q = (const float*)d_in[0];
    const float* k = (const float*)d_in[1];
    float* out = (float*)d_out;
    // blocks: B * (S/16) * (S/16) = 8 * 64 * 64
    dim3 grid(Bsz * 64 * 64);
    dim3 block(256);
    l1attn_wmma_kernel<<<grid, block, 0, stream>>>(q, k, out);
}
